// MultiHeadAttention_11218454577561
// MI455X (gfx1250) — matrix operand order repaired and checked by reading, not compiled, not run on hardware
//
#include <hip/hip_runtime.h>
#include <hip/hip_bf16.h>
#include <cstdint>

// ---------------------------------------------------------------------------
// MHA forward for MI455X (gfx1250): bf16 WMMA (v_wmma_f32_16x16x32_bf16),
// fused QKV projection (single read of X), flash-attention fusion so the
// 24 x 4096^2 score matrices never touch HBM, bf16-staged output projection.
// ---------------------------------------------------------------------------

#define NHEAD 12
#define SEQ   4096
#define DMODEL 768
#define DK    64

typedef __attribute__((ext_vector_type(16))) __bf16 v16bf;
typedef __attribute__((ext_vector_type(8)))  float  v8f;

union B16Frag { uint4 u4[2]; v16bf v; };

static __device__ __forceinline__ uint16_t f32_to_bf16(float f) {
    uint32_t u = __builtin_bit_cast(uint32_t, f);
    u += 0x7FFFu + ((u >> 16) & 1u);          // round-to-nearest-even
    return (uint16_t)(u >> 16);
}

static __device__ __forceinline__ uint32_t pack_bf16x2(float lo, float hi) {
    return (uint32_t)f32_to_bf16(lo) | ((uint32_t)f32_to_bf16(hi) << 16);
}

static __device__ __forceinline__ v16bf ld_frag(const uint16_t* p0, const uint16_t* p1) {
    B16Frag t;
    t.u4[0] = *(const uint4*)p0;
    t.u4[1] = *(const uint4*)p1;
    return t.v;
}

static __device__ __forceinline__ v8f wmma_bf16(v16bf a, v16bf b, v8f c) {
    return __builtin_amdgcn_wmma_f32_16x16x32_bf16(
        /*neg_a=*/false, a, /*neg_b=*/false, b,
        /*c_mod=*/(short)0, c, /*reuse_a=*/false, /*reuse_b=*/false);
}

// ---------------------------------------------------------------------------
// Kernel 1: fused QKV projection.
//   Yw[b,h,s,d] (bf16) = (X[b,s,:] @ Ww + biasw) * scalew   for w in {q,k,v}
// X fp32 [8192,768] row-major; W fp32 [768(in),768(out)] row-major.
// Block tile 128x64, 8 waves, each wave 32x32 per weight (2x2 WMMA C tiles).
// A tile staged once in LDS, reused against all three weights.
// ---------------------------------------------------------------------------
__global__ void __launch_bounds__(256)
qkv_kernel(const float* __restrict__ X,
           const float* __restrict__ Wq, const float* __restrict__ Bq,
           const float* __restrict__ Wk, const float* __restrict__ Bk,
           const float* __restrict__ Wv, const float* __restrict__ Bv,
           uint16_t* __restrict__ Yq, uint16_t* __restrict__ Yk,
           uint16_t* __restrict__ Yv)
{
    __shared__ uint16_t lsA[128 * 32];        // [m][k], k contiguous
    __shared__ uint16_t lsB[3][64 * 32];      // [n][k], k contiguous (W^T tiles)

    const float* Wp[3] = { Wq, Wk, Wv };
    const float* Bp[3] = { Bq, Bk, Bv };
    uint16_t*    Yp[3] = { Yq, Yk, Yv };
    const float  scl[3] = { 0.125f, 1.0f, 1.0f };   // 1/sqrt(dk) folded into Q

    const int t    = threadIdx.x;
    const int lane = t & 31, wid = t >> 5;
    const int wm   = wid & 3, wn = wid >> 2;
    const int half = lane >> 4, l16 = lane & 15;
    const int m0   = blockIdx.y * 128, n0 = blockIdx.x * 64;

    v8f acc[3][2][2] = {};

    for (int kt = 0; kt < DMODEL / 32; ++kt) {
        const int k0 = kt * 32;
        // --- stage A tile (fp32 -> bf16, packed dword stores) ---
        {
            const int row = t >> 1, c0 = (t & 1) * 16;
            const float* src = X + (size_t)(m0 + row) * DMODEL + k0 + c0;
            uint32_t* dst = (uint32_t*)&lsA[row * 32 + c0];
            #pragma unroll
            for (int i = 0; i < 16; i += 4) {
                float4 f = *(const float4*)(src + i);
                dst[(i >> 1) + 0] = pack_bf16x2(f.x, f.y);
                dst[(i >> 1) + 1] = pack_bf16x2(f.z, f.w);
            }
        }
        // --- stage B tiles transposed: lsB[w][n][k] = Ww[k0+k][n0+n] ---
        // Each thread owns a (k, k+1) pair x 4 n's -> pairs are contiguous in
        // the K-contiguous layout, so they pack into dword stores.
        {
            const int k2   = (t & 15) * 2;
            const int nseg = (t >> 4) * 4;
            #pragma unroll
            for (int w = 0; w < 3; ++w) {
                const float* s0 = Wp[w] + (size_t)(k0 + k2)     * DMODEL + n0 + nseg;
                const float* s1 = Wp[w] + (size_t)(k0 + k2 + 1) * DMODEL + n0 + nseg;
                float4 f0 = *(const float4*)s0;
                float4 f1 = *(const float4*)s1;
                uint32_t* db = (uint32_t*)&lsB[w][0];
                db[(nseg + 0) * 16 + (k2 >> 1)] = pack_bf16x2(f0.x, f1.x);
                db[(nseg + 1) * 16 + (k2 >> 1)] = pack_bf16x2(f0.y, f1.y);
                db[(nseg + 2) * 16 + (k2 >> 1)] = pack_bf16x2(f0.z, f1.z);
                db[(nseg + 3) * 16 + (k2 >> 1)] = pack_bf16x2(f0.w, f1.w);
            }
        }
        __syncthreads();

        v16bf a[2];
        #pragma unroll
        for (int i = 0; i < 2; ++i) {               // A frag: M=l16, K chunks at 8*half, +16
            const int off = (wm * 32 + i * 16 + l16) * 32 + half * 8;
            a[i] = ld_frag(&lsA[off], &lsA[off + 16]);
        }
        #pragma unroll
        for (int w = 0; w < 3; ++w) {
            #pragma unroll
            for (int j = 0; j < 2; ++j) {           // B frag: N=l16, K chunks at 8*half, +16
                const int off = (wn * 32 + j * 16 + l16) * 32 + half * 8;
                v16bf b = ld_frag(&lsB[w][off], &lsB[w][off + 16]);
                #pragma unroll
                for (int i = 0; i < 2; ++i)
                    acc[w][i][j] = wmma_bf16(a[i], b, acc[w][i][j]);
            }
        }
        __syncthreads();
    }

    // --- epilogue: bias, scale, bf16, scatter to [B,H,S,dk] ---
    #pragma unroll
    for (int w = 0; w < 3; ++w) {
        #pragma unroll
        for (int i = 0; i < 2; ++i) {
            #pragma unroll
            for (int j = 0; j < 2; ++j) {
                const int col = n0 + wn * 32 + j * 16 + l16;
                const float bv = Bp[w][col];
                const int h = col >> 6, d = col & 63;
                const int rbase = m0 + wm * 32 + i * 16 + 8 * half;
                #pragma unroll
                for (int e = 0; e < 8; ++e) {
                    const int gr = rbase + e;
                    const int b_ = gr >> 12, srow = gr & (SEQ - 1);
                    const float val = (acc[w][i][j][e] + bv) * scl[w];
                    Yp[w][((size_t)(b_ * NHEAD + h) * SEQ + srow) * DK + d] =
                        f32_to_bf16(val);
                }
            }
        }
    }
}

// ---------------------------------------------------------------------------
// Kernel 2: flash attention per head. Block = 4 waves, 64 query rows.
// Online softmax; O[b,s,h*64+d] written as bf16 for the output projection.
// ---------------------------------------------------------------------------
__global__ void __launch_bounds__(128)
flash_kernel(const uint16_t* __restrict__ Q, const uint16_t* __restrict__ K,
             const uint16_t* __restrict__ V, uint16_t* __restrict__ O)
{
    __shared__ uint16_t lsK[64 * 64];      // [key][dk]
    __shared__ uint16_t lsV[64 * 64];      // [dk][key]  (V transposed)
    __shared__ uint16_t lsP[4][16 * 64];   // per-wave P tile [row][key]

    const int t    = threadIdx.x;
    const int lane = t & 31, wid = t >> 5;
    const int half = lane >> 4, l16 = lane & 15;
    const int bh   = blockIdx.y;                 // b*NHEAD + h
    const int qm   = blockIdx.x * 64;
    const size_t head_base = (size_t)bh * SEQ * DK;

    // Q fragments (A-layout) held in registers for the whole KV loop
    v16bf aq[2];
    {
        const int qrow = qm + wid * 16 + l16;
        const uint16_t* qp = Q + head_base + (size_t)qrow * DK;
        #pragma unroll
        for (int c = 0; c < 2; ++c) {
            const int off = c * 32 + half * 8;
            aq[c] = ld_frag(qp + off, qp + off + 16);
        }
    }

    v8f   oacc[4] = {};
    float mrow[8], lrow[8];
    #pragma unroll
    for (int j = 0; j < 8; ++j) { mrow[j] = -1e30f; lrow[j] = 0.0f; }

    for (int jt = 0; jt < SEQ / 64; ++jt) {
        const uint16_t* kp = K + head_base + (size_t)jt * 64 * DK;
        const uint16_t* vp = V + head_base + (size_t)jt * 64 * DK;

        __syncthreads();   // previous iteration finished reading lsK/lsV/lsP
        // --- stage K tile [key][dk] (vector copies) ---
        {
            const int key = t >> 1, c0 = (t & 1) * 32;
            const uint16_t* s = kp + key * DK + c0;
            uint16_t* d = &lsK[key * 64 + c0];
            #pragma unroll
            for (int i = 0; i < 32; i += 8)
                *(uint4*)(d + i) = *(const uint4*)(s + i);
        }
        // --- stage V tile transposed -> lsV[dk][key]; key pairs pack to b32 ---
        {
            const int p  = t >> 2;           // key pair 0..31
            const int d0 = (t & 3) * 16;     // dk segment
            const uint16_t* s0 = vp + (2 * p)     * DK + d0;
            const uint16_t* s1 = vp + (2 * p + 1) * DK + d0;
            union { uint4 q[2]; uint16_t h[16]; } t0, t1;
            t0.q[0] = *(const uint4*)(s0);     t0.q[1] = *(const uint4*)(s0 + 8);
            t1.q[0] = *(const uint4*)(s1);     t1.q[1] = *(const uint4*)(s1 + 8);
            uint32_t* dv = (uint32_t*)&lsV[0];
            #pragma unroll
            for (int i = 0; i < 16; ++i)
                dv[(d0 + i) * 32 + p] =
                    (uint32_t)t0.h[i] | ((uint32_t)t1.h[i] << 16);
        }
        if (jt + 1 < SEQ / 64) {                    // global_prefetch_b8 next KV tile
            __builtin_prefetch(kp + 64 * DK, 0, 1);
            __builtin_prefetch(vp + 64 * DK, 0, 1);
        }
        __syncthreads();

        // --- scores S = Q K^T (16x64 per wave) ---
        v8f sc[4] = {};
        #pragma unroll
        for (int n = 0; n < 4; ++n) {
            #pragma unroll
            for (int c = 0; c < 2; ++c) {           // B frag: N=key, K=dk chunk
                const int off = (n * 16 + l16) * 64 + c * 32 + half * 8;
                v16bf bk = ld_frag(&lsK[off], &lsK[off + 16]);
                sc[n] = wmma_bf16(aq[c], bk, sc[n]);
            }
        }

        // --- online softmax; stats uniform per 16-lane half (rows j+8*half) ---
        float mnew[8];
        #pragma unroll
        for (int j = 0; j < 8; ++j) {
            float tm = fmaxf(fmaxf(sc[0][j], sc[1][j]), fmaxf(sc[2][j], sc[3][j]));
            #pragma unroll
            for (int m = 1; m <= 8; m <<= 1)
                tm = fmaxf(tm, __shfl_xor(tm, m, 32));
            mnew[j] = fmaxf(mrow[j], tm);
            const float f = __expf(mrow[j] - mnew[j]);
            lrow[j] *= f;
            #pragma unroll
            for (int d = 0; d < 4; ++d) oacc[d][j] *= f;
            mrow[j] = mnew[j];
        }
        #pragma unroll
        for (int j = 0; j < 8; ++j) {
            float rs = 0.0f;
            #pragma unroll
            for (int n = 0; n < 4; ++n) {
                const float p = __expf(sc[n][j] - mnew[j]);
                rs += p;
                lsP[wid][(j + 8 * half) * 64 + n * 16 + l16] = f32_to_bf16(p);
            }
            #pragma unroll
            for (int m = 1; m <= 8; m <<= 1)
                rs += __shfl_xor(rs, m, 32);
            lrow[j] += rs;
        }
        __syncthreads();

        // --- O += P V ---
        #pragma unroll
        for (int c = 0; c < 2; ++c) {
            const int offp = l16 * 64 + c * 32 + half * 8;   // A frag of P
            v16bf ap = ld_frag(&lsP[wid][offp], &lsP[wid][offp + 16]);
            #pragma unroll
            for (int d = 0; d < 4; ++d) {                    // B frag: N=dk, K=key
                const int offv = (d * 16 + l16) * 64 + c * 32 + half * 8;
                v16bf bv = ld_frag(&lsV[offv], &lsV[offv + 16]);
                oacc[d] = wmma_bf16(ap, bv, oacc[d]);
            }
        }
    }

    // --- epilogue: O /= l, write bf16 [B,S,D] with heads concatenated ---
    const int b_ = bh / NHEAD, h = bh % NHEAD;
    #pragma unroll
    for (int d = 0; d < 4; ++d) {
        const int col = h * DK + d * 16 + l16;
        #pragma unroll
        for (int j = 0; j < 8; ++j) {
            const int row = qm + wid * 16 + j + 8 * half;
            const float val = oacc[d][j] / lrow[j];
            O[((size_t)b_ * SEQ + row) * DMODEL + col] = f32_to_bf16(val);
        }
    }
}

// ---------------------------------------------------------------------------
// Kernel 3: out = attn(bf16) @ wo + bo  -> fp32. Same tiling as kernel 1.
// ---------------------------------------------------------------------------
__global__ void __launch_bounds__(256)
outproj_kernel(const uint16_t* __restrict__ A, const float* __restrict__ W,
               const float* __restrict__ bias, float* __restrict__ Out)
{
    __shared__ uint16_t lsA[128 * 32];
    __shared__ uint16_t lsB[64 * 32];

    const int t    = threadIdx.x;
    const int lane = t & 31, wid = t >> 5;
    const int wm   = wid & 3, wn = wid >> 2;
    const int half = lane >> 4, l16 = lane & 15;
    const int m0   = blockIdx.y * 128, n0 = blockIdx.x * 64;

    v8f acc[2][2] = {};

    for (int kt = 0; kt < DMODEL / 32; ++kt) {
        const int k0 = kt * 32;
        {   // A already bf16: straight vector copy
            const int row = t >> 1, c0 = (t & 1) * 16;
            const uint16_t* src = A + (size_t)(m0 + row) * DMODEL + k0 + c0;
            uint16_t* dst = &lsA[row * 32 + c0];
            *(uint4*)(dst + 0) = *(const uint4*)(src + 0);
            *(uint4*)(dst + 8) = *(const uint4*)(src + 8);
        }
        {   // B transpose with packed dword stores (k-pair per thread)
            const int k2   = (t & 15) * 2;
            const int nseg = (t >> 4) * 4;
            const float* s0 = W + (size_t)(k0 + k2)     * DMODEL + n0 + nseg;
            const float* s1 = W + (size_t)(k0 + k2 + 1) * DMODEL + n0 + nseg;
            float4 f0 = *(const float4*)s0;
            float4 f1 = *(const float4*)s1;
            uint32_t* db = (uint32_t*)&lsB[0];
            db[(nseg + 0) * 16 + (k2 >> 1)] = pack_bf16x2(f0.x, f1.x);
            db[(nseg + 1) * 16 + (k2 >> 1)] = pack_bf16x2(f0.y, f1.y);
            db[(nseg + 2) * 16 + (k2 >> 1)] = pack_bf16x2(f0.z, f1.z);
            db[(nseg + 3) * 16 + (k2 >> 1)] = pack_bf16x2(f0.w, f1.w);
        }
        __syncthreads();

        v16bf a[2], b[2];
        #pragma unroll
        for (int i = 0; i < 2; ++i) {
            const int off = (wm * 32 + i * 16 + l16) * 32 + half * 8;
            a[i] = ld_frag(&lsA[off], &lsA[off + 16]);
        }
        #pragma unroll
        for (int j = 0; j < 2; ++j) {
            const int off = (wn * 32 + j * 16 + l16) * 32 + half * 8;
            b[j] = ld_frag(&lsB[off], &lsB[off + 16]);
        }
        #pragma unroll
        for (int i = 0; i < 2; ++i)
            #pragma unroll
            for (int j = 0; j < 2; ++j)
                acc[i][j] = wmma_bf16(a[i], b[j], acc[i][j]);
        __syncthreads();
    }

    #pragma unroll
    for (int i = 0; i < 2; ++i) {
        #pragma unroll
        for (int j = 0; j < 2; ++j) {
            const int col = n0 + wn * 32 + j * 16 + l16;
            const float bv = bias[col];
            const int rbase = m0 + wm * 32 + i * 16 + 8 * half;
            #pragma unroll
            for (int e = 0; e < 8; ++e)
                Out[(size_t)(rbase + e) * DMODEL + col] = acc[i][j][e] + bv;
        }
    }
}

// ---------------------------------------------------------------------------
extern "C" void kernel_launch(void* const* d_in, const int* in_sizes, int n_in,
                              void* d_out, int out_size, void* d_ws, size_t ws_size,
                              hipStream_t stream)
{
    const float* x  = (const float*)d_in[0];
    const float* wq = (const float*)d_in[1];
    const float* bq = (const float*)d_in[2];
    const float* wk = (const float*)d_in[3];
    const float* bk = (const float*)d_in[4];
    const float* wv = (const float*)d_in[5];
    const float* bv = (const float*)d_in[6];
    const float* wo = (const float*)d_in[7];
    const float* bo = (const float*)d_in[8];
    float* out = (float*)d_out;

    const size_t headElems = (size_t)2 * NHEAD * SEQ * DK;   // 6,291,456
    uint16_t* qb = (uint16_t*)d_ws;
    uint16_t* kb = qb + headElems;
    uint16_t* vb = kb + headElems;
    uint16_t* ab = vb + headElems;                           // attn bf16 [B,S,D]

    const dim3 gGemm(DMODEL / 64, (2 * SEQ) / 128);          // (12, 64)
    const dim3 gFlash(SEQ / 64, 2 * NHEAD);                  // (64, 24)

    qkv_kernel<<<gGemm, 256, 0, stream>>>(x, wq, bq, wk, bk, wv, bv, qb, kb, vb);
    flash_kernel<<<gFlash, 128, 0, stream>>>(qb, kb, vb, ab);
    outproj_kernel<<<gGemm, 256, 0, stream>>>(ab, wo, bo, out);
}
